// GanaOrg_87050397156125
// MI455X (gfx1250) — compile-verified, matrix operand order checked by reading, not run on hardware
//
#include <hip/hip_runtime.h>
#include <hip/hip_bf16.h>
#include <math.h>

typedef __attribute__((ext_vector_type(16))) _Float16 v16h;
typedef __attribute__((ext_vector_type(8)))  float    v8f;

#define Nn 50000
#define En 800000
#define Fc 128
#define NF (Nn * Fc)
#define HASH_CAP (1u << 21)

// ---------------- helpers: monotone float<->uint for atomic max ------------
__device__ __forceinline__ unsigned enc_f(float f) {
  unsigned u = __float_as_uint(f);
  return (u & 0x80000000u) ? ~u : (u | 0x80000000u);
}
__device__ __forceinline__ float dec_f(unsigned k) {
  return __uint_as_float((k & 0x80000000u) ? (k & 0x7FFFFFFFu) : ~k);
}

// ---------------- sparse graph kernels -------------------------------------
// deg[src] += (valid ? (w>0) : w)
__global__ __launch_bounds__(256) void k_deg(const int* __restrict__ src,
                                             const float* __restrict__ w,
                                             float* __restrict__ deg, int valid) {
  int e = blockIdx.x * 256 + threadIdx.x;
  if (e >= En) return;
  float v = valid ? (w[e] > 0.f ? 1.f : 0.f) : w[e];
  atomicAdd(&deg[src[e]], v);
}

// wn[e] = dinv[src]*w*dinv[dst], dinv = rsqrt(deg) masked
__global__ __launch_bounds__(256) void k_wn(const int* __restrict__ src,
                                            const int* __restrict__ dst,
                                            const float* __restrict__ w,
                                            const float* __restrict__ deg,
                                            float* __restrict__ wn) {
  int e = blockIdx.x * 256 + threadIdx.x;
  if (e >= En) return;
  float ds = deg[src[e]], dd = deg[dst[e]];
  float is = ds > 0.f ? rsqrtf(ds) : 0.f;
  float id = dd > 0.f ? rsqrtf(dd) : 0.f;
  wn[e] = is * w[e] * id;
}

// zout[dst,:] += scale * wn[e] * zin[src,:]   (4 features/thread, float4 load)
__global__ __launch_bounds__(256) void k_scatter(const int* __restrict__ src,
                                                 const int* __restrict__ dst,
                                                 const float* __restrict__ wn,
                                                 const float* __restrict__ zin,
                                                 float* __restrict__ zout,
                                                 float scale) {
  int idx = blockIdx.x * 256 + threadIdx.x;      // E*32 threads
  int e  = idx >> 5;
  int f0 = (idx & 31) << 2;
  if (e >= En) return;
  float c = scale * wn[e];
  const float4 v = *(const float4*)(zin + (size_t)src[e] * Fc + f0);
  float* o = zout + (size_t)dst[e] * Fc + f0;
  atomicAdd(o + 0, c * v.x);
  atomicAdd(o + 1, c * v.y);
  atomicAdd(o + 2, c * v.z);
  atomicAdd(o + 3, c * v.w);
}

// out[i] += a*x[i]
__global__ __launch_bounds__(256) void k_axpy(float* __restrict__ out, float a,
                                              const float* __restrict__ x, int n) {
  int i = blockIdx.x * 256 + threadIdx.x;
  if (i < n) out[i] += a * x[i];
}

// f32 -> f16
__global__ __launch_bounds__(256) void k_conv(_Float16* __restrict__ dst,
                                              const float* __restrict__ src, int n) {
  int i = blockIdx.x * 256 + threadIdx.x;
  if (i < n) dst[i] = (_Float16)src[i];
}

// Pre-swizzle W[K=3][128][128] (f32, row-major KxH) into per-lane WMMA
// B-fragment layout: whb[((k*4+kk)*8 + ntile)*32 + lane][j] ; 49152 halves.
__global__ __launch_bounds__(256) void k_swizzleW(const float* __restrict__ Wf,
                                                  _Float16* __restrict__ whb) {
  int idx = blockIdx.x * 256 + threadIdx.x;
  if (idx >= 3 * 4 * 8 * 32 * 16) return;
  int j    = idx & 15;
  int lane = (idx >> 4) & 31;
  int nt   = (idx >> 9) & 7;
  int kk   = (idx >> 12) & 3;
  int k    = idx >> 14;
  int kb   = (lane & 16) ? 8 : 0;
  int krel = kb + j + ((j >= 8) ? 8 : 0);      // ISA 16-bit A/B K pattern
  int kr   = kk * 32 + krel;
  int n    = nt * 16 + (lane & 15);
  whb[idx] = (_Float16)Wf[((size_t)k * Fc + kr) * Fc + n];
}

// ---------------- WMMA GEMM: out = ELU(Z0@W0 + Z1@W1 + Z2@W2 + b) ----------
__global__ __launch_bounds__(256) void k_cheb_gemm(
    const _Float16* __restrict__ z0, const _Float16* __restrict__ z1,
    const _Float16* __restrict__ z2, const _Float16* __restrict__ whb,
    const float* __restrict__ bias, float* __restrict__ out) {
  const int lane  = threadIdx.x & 31;
  const int wtile = threadIdx.x >> 5;          // n-tile 0..7 (H = 8*16)
  const int m0    = blockIdx.x * 16;
  const int mrow  = m0 + (lane & 15);
  const int kb    = (lane & 16) ? 8 : 0;
  const _Float16* zs[3] = { z0, z1, z2 };
  v8f acc = {};
  union AB { v16h v; float4 f[2]; };
#pragma unroll
  for (int k = 0; k < 3; ++k) {
#pragma unroll
    for (int kk = 0; kk < 4; ++kk) {
      AB a, b;
      const _Float16* pa = zs[k] + (size_t)mrow * Fc + kk * 32 + kb;
      a.f[0] = *(const float4*)(pa);
      a.f[1] = *(const float4*)(pa + 16);
      const _Float16* pb = whb + ((((k * 4 + kk) * 8 + wtile) * 32 + lane) << 4);
      b.f[0] = *(const float4*)(pb);
      b.f[1] = *(const float4*)(pb + 8);
      acc = __builtin_amdgcn_wmma_f32_16x16x32_f16(false, a.v, false, b.v,
                                                   (short)0, acc, false, false);
    }
  }
  const int n  = wtile * 16 + (lane & 15);
  const float bv = bias[n];
  const int mb = m0 + ((lane & 16) ? 8 : 0);
#pragma unroll
  for (int r = 0; r < 8; ++r) {                // VGPR r -> row mb+r, col n
    float v = acc[r] + bv;
    v = v > 0.f ? v : (expf(v) - 1.f);         // ELU
    out[(size_t)(mb + r) * Fc + n] = v;
  }
}

// ---------------- graclus pooling -------------------------------------------
__global__ __launch_bounds__(256) void k_cut(const int* __restrict__ src,
                                             const int* __restrict__ dst,
                                             const float* __restrict__ w,
                                             const float* __restrict__ deg,
                                             float* __restrict__ cut) {
  int e = blockIdx.x * 256 + threadIdx.x;
  if (e >= En) return;
  float we = w[e];
  if (we > 0.f) {
    float a = deg[src[e]], b = deg[dst[e]];
    cut[e] = we * ((a > 0.f ? 1.f / a : 0.f) + (b > 0.f ? 1.f / b : 0.f));
  } else {
    cut[e] = -INFINITY;
  }
}

__global__ __launch_bounds__(256) void k_segmax(const int* __restrict__ src,
                                                const float* __restrict__ cut,
                                                int* __restrict__ mbits) {
  int e = blockIdx.x * 256 + threadIdx.x;
  if (e >= En) return;
  float c = cut[e];
  if (c > -INFINITY) atomicMax(&mbits[src[e]], __float_as_int(c)); // c>0 -> int-ordered
}

__global__ __launch_bounds__(256) void k_partner(const int* __restrict__ src,
                                                 const int* __restrict__ dst,
                                                 const float* __restrict__ cut,
                                                 const int* __restrict__ mbits,
                                                 int* __restrict__ partner) {
  int e = blockIdx.x * 256 + threadIdx.x;
  if (e >= En) return;
  float c = cut[e];
  if (c > -INFINITY && __float_as_int(c) == mbits[src[e]])
    atomicMax(&partner[src[e]], dst[e]);
}

__global__ __launch_bounds__(256) void k_mutual(const int* __restrict__ partner,
                                                int* __restrict__ c) {
  int i = blockIdx.x * 256 + threadIdx.x;
  if (i >= Nn) return;
  int p = partner[i];
  int pc = min(max(p, 0), Nn - 1);
  bool mut = (p >= 0) && (p != i) && (partner[pc] == i);
  c[i] = mut ? min(i, p) : i;
}

// ---------------- pool_x (segment max) --------------------------------------
__global__ __launch_bounds__(256) void k_poolx_scatter(const int* __restrict__ c,
                                                       const float* __restrict__ x,
                                                       unsigned* __restrict__ xmax,
                                                       int* __restrict__ cnt) {
  int idx = blockIdx.x * 256 + threadIdx.x;    // N*32 threads
  int node = idx >> 5;
  int f0 = (idx & 31) << 2;
  if (node >= Nn) return;
  int cl = c[node];
  const float4 v = *(const float4*)(x + (size_t)node * Fc + f0);
  unsigned* o = xmax + (size_t)cl * Fc + f0;
  atomicMax(o + 0, enc_f(v.x));
  atomicMax(o + 1, enc_f(v.y));
  atomicMax(o + 2, enc_f(v.z));
  atomicMax(o + 3, enc_f(v.w));
  if (f0 == 0) atomicAdd(&cnt[cl], 1);
}

__global__ __launch_bounds__(256) void k_poolx_final(const unsigned* __restrict__ xmax,
                                                     const int* __restrict__ cnt,
                                                     float* __restrict__ xp) {
  int i = blockIdx.x * 256 + threadIdx.x;
  if (i >= NF) return;
  xp[i] = (cnt[i >> 7] > 0) ? dec_f(xmax[i]) : 0.f;
}

// ---------------- coalesce pooled edges (hash dedup) ------------------------
__global__ __launch_bounds__(256) void k_pool_edges(const int* __restrict__ src,
                                                    const int* __restrict__ dst,
                                                    const int* __restrict__ c,
                                                    int* __restrict__ s2,
                                                    int* __restrict__ d2,
                                                    float* __restrict__ w2,
                                                    unsigned* __restrict__ hash) {
  int e = blockIdx.x * 256 + threadIdx.x;
  if (e >= En) return;
  int s = c[src[e]], d = c[dst[e]];
  s2[e] = s; d2[e] = d;
  float w = 0.f;
  if (s != d) {
    unsigned key = (unsigned)s * 50000u + (unsigned)d;   // < 2.5e9, != EMPTY
    unsigned h = (key * 2654435761u) & (HASH_CAP - 1);
    for (;;) {
      unsigned old = atomicCAS(&hash[h], 0xFFFFFFFFu, key);
      if (old == 0xFFFFFFFFu) { w = 1.f; break; }        // first of this pair
      if (old == key) { w = 0.f; break; }                // duplicate
      h = (h + 1) & (HASH_CAP - 1);
    }
  }
  w2[e] = w;
}

// ---------------- gather + log_softmax --------------------------------------
__global__ __launch_bounds__(128) void k_lsm(const float* __restrict__ xp2,
                                             const int* __restrict__ c1,
                                             const int* __restrict__ c2,
                                             float* __restrict__ out) {
  __shared__ float s[128];
  int i = blockIdx.x, f = threadIdx.x;
  int g = c2[c1[i]];
  float v = xp2[(size_t)g * Fc + f];
  s[f] = v; __syncthreads();
  for (int o = 64; o > 0; o >>= 1) { if (f < o) s[f] = fmaxf(s[f], s[f + o]); __syncthreads(); }
  float mx = s[0]; __syncthreads();
  s[f] = expf(v - mx); __syncthreads();
  for (int o = 64; o > 0; o >>= 1) { if (f < o) s[f] += s[f + o]; __syncthreads(); }
  float lse = logf(s[0]) + mx;
  out[(size_t)i * Fc + f] = v - lse;
}

// ---------------- host-side helpers -----------------------------------------
static void cheb_block(const int* src, const int* dst, const float* w,
                       const float* xin, const float* Wf, const float* bias,
                       float* out, float* deg, float* wn, float* z1, float* z2,
                       _Float16* xh, _Float16* z1h, _Float16* z2h, _Float16* whb,
                       hipStream_t stream) {
  hipMemsetAsync(deg, 0, Nn * 4, stream);
  k_deg<<<3125, 256, 0, stream>>>(src, w, deg, 0);
  k_wn<<<3125, 256, 0, stream>>>(src, dst, w, deg, wn);
  hipMemsetAsync(z1, 0, (size_t)NF * 4, stream);
  k_scatter<<<100000, 256, 0, stream>>>(src, dst, wn, xin, z1, -1.f);   // Z1 = -L~ x
  hipMemsetAsync(z2, 0, (size_t)NF * 4, stream);
  k_scatter<<<100000, 256, 0, stream>>>(src, dst, wn, z1, z2, -2.f);    // Z2 = 2*(-L~ Z1)
  k_axpy<<<25000, 256, 0, stream>>>(z2, -1.f, xin, NF);                 //      - Z0
  k_conv<<<25000, 256, 0, stream>>>(xh, xin, NF);
  k_conv<<<25000, 256, 0, stream>>>(z1h, z1, NF);
  k_conv<<<25000, 256, 0, stream>>>(z2h, z2, NF);
  k_swizzleW<<<192, 256, 0, stream>>>(Wf, whb);
  k_cheb_gemm<<<3125, 256, 0, stream>>>(xh, z1h, z2h, whb, bias, out);
}

static void graclus(const int* src, const int* dst, const float* w,
                    float* deg, float* cut, int* mbits, int* partner, int* c,
                    hipStream_t stream) {
  hipMemsetAsync(deg, 0, Nn * 4, stream);
  k_deg<<<3125, 256, 0, stream>>>(src, w, deg, 1);
  k_cut<<<3125, 256, 0, stream>>>(src, dst, w, deg, cut);
  hipMemsetAsync(mbits, 0, Nn * 4, stream);                 // 0 < any positive cut bits
  k_segmax<<<3125, 256, 0, stream>>>(src, cut, mbits);
  hipMemsetAsync(partner, 0xFF, Nn * 4, stream);            // -1
  k_partner<<<3125, 256, 0, stream>>>(src, dst, cut, mbits, partner);
  k_mutual<<<196, 256, 0, stream>>>(partner, c);
}

static void pool_x(const int* c, const float* x, unsigned* xmax, int* cnt,
                   float* xp, hipStream_t stream) {
  hipMemsetAsync(xmax, 0, (size_t)NF * 4, stream);          // enc key 0 == "-inf"
  hipMemsetAsync(cnt, 0, Nn * 4, stream);
  k_poolx_scatter<<<6250, 256, 0, stream>>>(c, x, xmax, cnt);
  k_poolx_final<<<25000, 256, 0, stream>>>(xmax, cnt, xp);
}

extern "C" void kernel_launch(void* const* d_in, const int* in_sizes, int n_in,
                              void* d_out, int out_size, void* d_ws, size_t ws_size,
                              hipStream_t stream) {
  const float* x  = (const float*)d_in[0];
  const float* ew = (const float*)d_in[1];
  const float* W1 = (const float*)d_in[2];
  const float* b1 = (const float*)d_in[3];
  const float* W2 = (const float*)d_in[4];
  const float* b2 = (const float*)d_in[5];
  const int* src  = (const int*)d_in[6];
  const int* dst  = src + En;
  float* out = (float*)d_out;

  // carve workspace (256B-aligned slabs)
  char* base = (char*)d_ws;
  auto alloc = [&](size_t bytes) { char* p = base; base += (bytes + 255) & ~(size_t)255; return p; };
  float*     deg  = (float*)alloc((size_t)Nn * 4);
  float*     wn   = (float*)alloc((size_t)En * 4);          // also reused as `cut`
  float*     z1   = (float*)alloc((size_t)NF * 4);          // also xmax scratch
  float*     z2   = (float*)alloc((size_t)NF * 4);          // also xp2
  float*     x1   = (float*)alloc((size_t)NF * 4);          // block1 out
  float*     xp1  = (float*)alloc((size_t)NF * 4);
  float*     x2   = (float*)alloc((size_t)NF * 4);
  _Float16*  xh   = (_Float16*)alloc((size_t)NF * 2);
  _Float16*  z1h  = (_Float16*)alloc((size_t)NF * 2);
  _Float16*  z2h  = (_Float16*)alloc((size_t)NF * 2);
  _Float16*  whb  = (_Float16*)alloc((size_t)3 * 4 * 8 * 32 * 16 * 2);
  int*       mbits   = (int*)alloc((size_t)Nn * 4);
  int*       partner = (int*)alloc((size_t)Nn * 4);
  int*       c1      = (int*)alloc((size_t)Nn * 4);
  int*       c2      = (int*)alloc((size_t)Nn * 4);
  int*       cnt     = (int*)alloc((size_t)Nn * 4);
  int*       s2      = (int*)alloc((size_t)En * 4);
  int*       d2      = (int*)alloc((size_t)En * 4);
  float*     w2      = (float*)alloc((size_t)En * 4);
  unsigned*  hash    = (unsigned*)alloc((size_t)HASH_CAP * 4);

  // ---- block 1: ChebConv -> ELU -------------------------------------------
  cheb_block(src, dst, ew, x, W1, b1, x1, deg, wn, z1, z2, xh, z1h, z2h, whb, stream);
  // graclus on original graph
  graclus(src, dst, ew, deg, wn, mbits, partner, c1, stream);
  // pool features (xmax scratch aliases z1 — free after GEMM)
  pool_x(c1, x1, (unsigned*)z1, cnt, xp1, stream);
  // coalesce edges
  hipMemsetAsync(hash, 0xFF, (size_t)HASH_CAP * 4, stream);
  k_pool_edges<<<3125, 256, 0, stream>>>(src, dst, c1, s2, d2, w2, hash);

  // ---- block 2 on pooled graph --------------------------------------------
  cheb_block(s2, d2, w2, xp1, W2, b2, x2, deg, wn, z1, z2, xh, z1h, z2h, whb, stream);
  graclus(s2, d2, w2, deg, wn, mbits, partner, c2, stream);
  float* xp2 = z2;                                          // z2 free after GEMM2
  pool_x(c2, x2, (unsigned*)z1, cnt, xp2, stream);

  // ---- final gather + log_softmax -----------------------------------------
  k_lsm<<<Nn, 128, 0, stream>>>(xp2, c1, c2, out);
}
